// CF_67104569033471
// MI455X (gfx1250) — matrix operand order repaired and checked by reading, not compiled, not run on hardware
//
#include <hip/hip_runtime.h>
#include <math.h>

// ---------------------------------------------------------------------------
// MI455X (gfx1250) implementation.
//   - All GEMMs via v_wmma_f32_16x16x32_bf16 with bf16 hi/lo split (bf16x3):
//     C = Ahi*Bhi + Ahi*Blo + Alo*Bhi  ->  ~f32 accuracy at bf16 WMMA rate.
//   - Each wave computes a 16x32 tile (2 N-tiles) reusing A fragments; B
//     fragments for step k+1 are prefetched before the k-step WMMAs.
//   - Write path: softmax column sums cancel in w = sq[n,t]/colmax[t], so the
//     big [N x 430] score matrices are never materialized; per-row max/argmax
//     and per-column max are fused into the score GEMM epilogue.
// ---------------------------------------------------------------------------

typedef __attribute__((ext_vector_type(16))) __bf16 v16bf;
typedef __attribute__((ext_vector_type(8)))  float  v8f;

#define GEMM_ROWS    64     // rows of A per workgroup (4 waves x 16-row tiles)
#define GEMM_THREADS 128
#define CDIV(a, b) (((a) + (b) - 1) / (b))

struct B32x2 { uint4 a, b; };

__device__ __forceinline__ v16bf mk16(uint4 x, uint4 y) {
  B32x2 t{x, y};
  return __builtin_bit_cast(v16bf, t);
}

// monotonic float <-> ordered-uint mapping (for atomic float max via uint max)
__device__ __forceinline__ unsigned encf(float f) {
  unsigned b = __float_as_uint(f);
  return b ^ ((unsigned)((int)b >> 31) | 0x80000000u);
}
__device__ __forceinline__ float decf(unsigned u) {
  unsigned b = (u & 0x80000000u) ? (u ^ 0x80000000u) : ~u;
  return __uint_as_float(b);
}

__device__ __forceinline__ float waveRedAdd(float v) {
  #pragma unroll
  for (int m = 16; m >= 1; m >>= 1) v += __shfl_xor(v, m);
  return v;
}
__device__ __forceinline__ float waveRedMax(float v) {
  #pragma unroll
  for (int m = 16; m >= 1; m >>= 1) v = fmaxf(v, __shfl_xor(v, m));
  return v;
}

// ---------------------------------------------------------------------------
// f32 -> bf16 hi/lo split with zero padding; transp=1 reads in[c][r].
// out dims [Rout][Cout]; in dims [Rin][Cin].
// ---------------------------------------------------------------------------
__global__ void k_split(const float* __restrict__ in, int Rin, int Cin,
                        __bf16* __restrict__ hi, __bf16* __restrict__ lo,
                        int Rout, int Cout, int transp) {
  int i = blockIdx.x * blockDim.x + threadIdx.x;
  if (i >= Rout * Cout) return;
  int r = i / Cout, c = i - r * Cout;
  float x = 0.f;
  if (!transp) { if (r < Rin && c < Cin) x = in[(size_t)r * Cin + c]; }
  else         { if (c < Rin && r < Cin) x = in[(size_t)c * Cin + r]; }
  __bf16 h = (__bf16)x;
  hi[i] = h;
  lo[i] = (__bf16)(x - (float)h);
}

// one wave per 512-wide row: rn = 1 / max(||row||, 1e-12)
__global__ void k_rownorm(const float* __restrict__ A, float* __restrict__ rn) {
  int row  = blockIdx.x * 8 + (threadIdx.x >> 5);
  int lane = threadIdx.x & 31;
  const float4* p = (const float4*)(A + (size_t)row * 512);
  float s = 0.f;
  #pragma unroll
  for (int i = 0; i < 4; ++i) {
    float4 x = p[lane + i * 32];
    s += x.x * x.x + x.y * x.y + x.z * x.z + x.w * x.w;
  }
  s = waveRedAdd(s);
  if (lane == 0) rn[row] = 1.0f / fmaxf(sqrtf(s), 1e-12f);
}

// copy text_token [768][512] into Acat[:, 0:512] (stride 1024)
__global__ void k_copy_text(const float* __restrict__ src, float* __restrict__ Acat) {
  int i = blockIdx.x * blockDim.x + threadIdx.x;   // over 768*512
  int r = i >> 9, c = i & 511;
  Acat[(size_t)r * 1024 + c] = src[i];
}

// ---------------------------------------------------------------------------
// Generic bf16x3 WMMA GEMM:  C[M x N] = (A .* rnormA) @ BT^T
//   A:  f32 [M][lda], staged in LDS as bf16 hi/lo (row-scaled by rnormA).
//   BT: bf16 hi/lo [Ntiles*16][ldb]  (BT[n][k] contiguous in k, zero-padded).
//   Ntiles must be even (waves process pairs of 16-col tiles).
//   mode 0: store C
//   mode 1: C = alpha*acc + resid   (extractor)
//   mode 2: fused per-row (max, argmax) + per-column max (ordered-uint atomics)
// ---------------------------------------------------------------------------
__global__ void __launch_bounds__(GEMM_THREADS)
k_gemm(const float* __restrict__ A, int lda, const float* __restrict__ rnormA,
       const __bf16* __restrict__ BT_hi, const __bf16* __restrict__ BT_lo, int ldb,
       int K, int Ntiles, int Nreal,
       float* __restrict__ C, int ldc,
       const float* __restrict__ resid, float alpha,
       float* __restrict__ rowmax_out, int* __restrict__ rowarg_out,
       unsigned* __restrict__ colmax_u, int mode) {
  extern __shared__ char smem[];
  __bf16*   sh_hi   = (__bf16*)smem;
  __bf16*   sh_lo   = sh_hi + GEMM_ROWS * K;
  unsigned* sh_cmax = (unsigned*)(sh_lo + GEMM_ROWS * K);

  const int tid   = threadIdx.x;
  const int Mbase = blockIdx.x * GEMM_ROWS;

  // stage A tile (64 x K) as bf16 hi/lo, row-normalized
  for (int idx = tid * 4; idx < GEMM_ROWS * K; idx += GEMM_THREADS * 4) {
    int row = idx / K, col = idx - row * K;
    float4 x = *(const float4*)(A + (size_t)(Mbase + row) * lda + col);
    float s = rnormA ? rnormA[Mbase + row] : 1.0f;
    float v[4] = {x.x * s, x.y * s, x.z * s, x.w * s};
    #pragma unroll
    for (int e = 0; e < 4; ++e) {
      __bf16 h = (__bf16)v[e];
      sh_hi[idx + e] = h;
      sh_lo[idx + e] = (__bf16)(v[e] - (float)h);
    }
  }
  if (mode == 2)
    for (int i = tid; i < Ntiles * 16; i += GEMM_THREADS) sh_cmax[i] = 0u;
  __syncthreads();

  const int lane   = tid & 31;
  const int wave   = tid >> 5;
  const int hs     = (lane >> 4) & 1;    // lane-half select
  const int lcol   = lane & 15;
  const int Ktiles = K >> 5;
  const int Npairs = Ntiles >> 1;

  // per-lane A fragment base (contiguous K order matching B: lanes 0-15 hold K{0-15},
  // lanes 16-31 hold K{16-31})
  const __bf16* ah = sh_hi + (size_t)(wave * 16 + lcol) * K;
  const __bf16* al = sh_lo + (size_t)(wave * 16 + lcol) * K;
  const int o1 = hs ? 16 : 0;
  const int o2 = hs ? 24 : 8;

  float rbest[8];
  int   rarg[8];
  #pragma unroll
  for (int r = 0; r < 8; ++r) { rbest[r] = -INFINITY; rarg[r] = 0; }

  for (int jj = 0; jj < Npairs; ++jj) {
    v8f acc0 = {0.f, 0.f, 0.f, 0.f, 0.f, 0.f, 0.f, 0.f};
    v8f acc1 = {0.f, 0.f, 0.f, 0.f, 0.f, 0.f, 0.f, 0.f};
    // B layout: lane n (=lcol) holds 16 contiguous K; lanes 16-31 take K+16
    const __bf16* b0h = BT_hi + (size_t)(jj * 32 + lcol) * ldb + (hs ? 16 : 0);
    const __bf16* b0l = BT_lo + (size_t)(jj * 32 + lcol) * ldb + (hs ? 16 : 0);
    const __bf16* b1h = b0h + (size_t)16 * ldb;
    const __bf16* b1l = b0l + (size_t)16 * ldb;

    // prefetch k = 0 B fragments
    v16bf bH0 = mk16(*(const uint4*)(b0h), *(const uint4*)(b0h + 8));
    v16bf bL0 = mk16(*(const uint4*)(b0l), *(const uint4*)(b0l + 8));
    v16bf bH1 = mk16(*(const uint4*)(b1h), *(const uint4*)(b1h + 8));
    v16bf bL1 = mk16(*(const uint4*)(b1l), *(const uint4*)(b1l + 8));

    for (int k = 0; k < Ktiles; ++k) {
      const int kk = k * 32;
      v16bf aH = mk16(*(const uint4*)(ah + kk + o1), *(const uint4*)(ah + kk + o2));
      v16bf aL = mk16(*(const uint4*)(al + kk + o1), *(const uint4*)(al + kk + o2));
      // prefetch k+1 B fragments (clamped on last step; avoids divergence)
      const int kn = (k + 1 < Ktiles) ? kk + 32 : kk;
      v16bf nH0 = mk16(*(const uint4*)(b0h + kn), *(const uint4*)(b0h + kn + 8));
      v16bf nL0 = mk16(*(const uint4*)(b0l + kn), *(const uint4*)(b0l + kn + 8));
      v16bf nH1 = mk16(*(const uint4*)(b1h + kn), *(const uint4*)(b1h + kn + 8));
      v16bf nL1 = mk16(*(const uint4*)(b1l + kn), *(const uint4*)(b1l + kn + 8));

      acc0 = __builtin_amdgcn_wmma_f32_16x16x32_bf16(false, aH, false, bH0, (short)0, acc0, false, false);
      acc1 = __builtin_amdgcn_wmma_f32_16x16x32_bf16(false, aH, false, bH1, (short)0, acc1, false, false);
      acc0 = __builtin_amdgcn_wmma_f32_16x16x32_bf16(false, aH, false, bL0, (short)0, acc0, false, false);
      acc1 = __builtin_amdgcn_wmma_f32_16x16x32_bf16(false, aH, false, bL1, (short)0, acc1, false, false);
      acc0 = __builtin_amdgcn_wmma_f32_16x16x32_bf16(false, aL, false, bH0, (short)0, acc0, false, false);
      acc1 = __builtin_amdgcn_wmma_f32_16x16x32_bf16(false, aL, false, bH1, (short)0, acc1, false, false);

      bH0 = nH0; bL0 = nL0; bH1 = nH1; bL1 = nL1;
    }

    #pragma unroll
    for (int t = 0; t < 2; ++t) {
      const v8f& acc = t ? acc1 : acc0;
      const int colg = jj * 32 + t * 16 + lcol;
      if (mode == 0) {
        #pragma unroll
        for (int r = 0; r < 8; ++r) {
          int rowg = Mbase + wave * 16 + (hs ? r + 8 : r);
          C[(size_t)rowg * ldc + colg] = acc[r];
        }
      } else if (mode == 1) {
        #pragma unroll
        for (int r = 0; r < 8; ++r) {
          int rowg = Mbase + wave * 16 + (hs ? r + 8 : r);
          C[(size_t)rowg * ldc + colg] = alpha * acc[r] + resid[(size_t)rowg * ldc + colg];
        }
      } else {
        const bool valid = colg < Nreal;
        float cm = -INFINITY;
        #pragma unroll
        for (int r = 0; r < 8; ++r) {
          float v = valid ? acc[r] : -INFINITY;
          if (v > rbest[r]) { rbest[r] = v; rarg[r] = colg; }  // first-index ties
          cm = fmaxf(cm, v);
        }
        cm = fmaxf(cm, __shfl_xor(cm, 16));   // both halves share the column
        if (valid && lane < 16) atomicMax(&sh_cmax[colg], encf(cm));
      }
    }
  }

  if (mode == 2) {
    #pragma unroll
    for (int r = 0; r < 8; ++r) {
      float v = rbest[r];
      int   a = rarg[r];
      #pragma unroll
      for (int m = 8; m >= 1; m >>= 1) {
        float vo = __shfl_xor(v, m);
        int   ao = __shfl_xor(a, m);
        if (vo > v || (vo == v && ao < a)) { v = vo; a = ao; }
      }
      if (lcol == 0) {
        int rowg = Mbase + wave * 16 + (hs ? r + 8 : r);
        rowmax_out[rowg] = v;
        rowarg_out[rowg] = a;
      }
    }
    __syncthreads();
    for (int i = tid; i < Nreal; i += GEMM_THREADS)
      atomicMax(&colmax_u[i], sh_cmax[i]);
  }
}

// row softmax over 430 valid cols of score[768][448] -> P[768][448] (zero pad)
__global__ void k_softmax(const float* __restrict__ score, float* __restrict__ P) {
  int row  = blockIdx.x * 8 + (threadIdx.x >> 5);
  int lane = threadIdx.x & 31;
  float v[14];
  float mx = -INFINITY;
  #pragma unroll
  for (int i = 0; i < 14; ++i) {
    int c = lane + i * 32;
    float x = (c < 430) ? score[(size_t)row * 448 + c] : -INFINITY;
    v[i] = x;
    mx = fmaxf(mx, x);
  }
  mx = waveRedMax(mx);
  float s = 0.f;
  #pragma unroll
  for (int i = 0; i < 14; ++i) {
    int c = lane + i * 32;
    float e = (c < 430) ? expf(v[i] - mx) : 0.f;
    v[i] = e;
    s += e;
  }
  s = waveRedAdd(s);
  float inv = 1.0f / s;
  #pragma unroll
  for (int i = 0; i < 14; ++i) {
    int c = lane + i * 32;                // covers 0..447
    P[(size_t)row * 448 + c] = v[i] * inv;
  }
}

// per-row scatter-reduce: w[n] = exp(s_n - colmax[t]); sums[t] += w*base[n]
__global__ void k_scatter(const float* __restrict__ img, const float* __restrict__ rnorm,
                          const float* __restrict__ rowmax, const int* __restrict__ rowarg,
                          const unsigned* __restrict__ colmax_u,
                          float* __restrict__ sums, float* __restrict__ cnt, int Nrows) {
  int row = blockIdx.x * 8 + (threadIdx.x >> 5);
  if (row >= Nrows) return;
  int lane = threadIdx.x & 31;
  int t    = rowarg[row];
  float w  = expf(rowmax[row] - decf(colmax_u[t]));
  float s  = w * rnorm[row];
  const float4* p = (const float4*)(img + (size_t)row * 512);
  float* dst = sums + (size_t)t * 512;
  #pragma unroll
  for (int i = 0; i < 4; ++i) {
    int c4 = lane + i * 32;
    float4 x = p[c4];
    atomicAdd(dst + c4 * 4 + 0, s * x.x);
    atomicAdd(dst + c4 * 4 + 1, s * x.y);
    atomicAdd(dst + c4 * 4 + 2, s * x.z);
    atomicAdd(dst + c4 * 4 + 3, s * x.w);
  }
  if (lane == 0) atomicAdd(&cnt[t], 1.0f);
}

// new_cache[m] = mean over 3 scales of l2norm(cnt>0 ? .8*cache+.2*sums : cache)
__global__ void k_combine(const float* __restrict__ cache, const float* __restrict__ sums,
                          const float* __restrict__ cnt, float* __restrict__ out) {
  int m = blockIdx.x * 8 + (threadIdx.x >> 5);
  if (m >= 430) return;
  int lane = threadIdx.x & 31;
  float acc[16];
  #pragma unroll
  for (int i = 0; i < 16; ++i) acc[i] = 0.f;
  for (int sc = 0; sc < 3; ++sc) {
    float c = cnt[sc * 430 + m];
    const float* su = sums + ((size_t)sc * 430 + m) * 512;
    const float* ca = cache + (size_t)m * 512;
    float u[16], ss = 0.f;
    #pragma unroll
    for (int i = 0; i < 16; ++i) {
      int idx = lane + i * 32;
      float v = (c > 0.f) ? (0.8f * ca[idx] + 0.2f * su[idx]) : ca[idx];
      u[i] = v;
      ss += v * v;
    }
    ss = waveRedAdd(ss);
    float rn = 1.0f / fmaxf(sqrtf(ss), 1e-12f);
    #pragma unroll
    for (int i = 0; i < 16; ++i) acc[i] += u[i] * rn;
  }
  #pragma unroll
  for (int i = 0; i < 16; ++i)
    out[(size_t)m * 512 + lane + i * 32] = acc[i] * (1.0f / 3.0f);
}

// loss = mean |l2norm(text_fine) - text_token|
__global__ void k_loss(const float* __restrict__ tf, const float* __restrict__ text,
                       float* __restrict__ loss) {
  int row  = blockIdx.x * 8 + (threadIdx.x >> 5);
  int lane = threadIdx.x & 31;
  float x[16], ss = 0.f;
  #pragma unroll
  for (int i = 0; i < 16; ++i) {
    x[i] = tf[(size_t)row * 512 + lane + i * 32];
    ss += x[i] * x[i];
  }
  ss = waveRedAdd(ss);
  float rn = 1.0f / fmaxf(sqrtf(ss), 1e-12f);
  float a = 0.f;
  #pragma unroll
  for (int i = 0; i < 16; ++i)
    a += fabsf(x[i] * rn - text[(size_t)row * 512 + lane + i * 32]);
  a = waveRedAdd(a);
  if (lane == 0) atomicAdd(loss, a * (1.0f / (768.0f * 512.0f)));
}

// ---------------------------------------------------------------------------
extern "C" void kernel_launch(void* const* d_in, const int* in_sizes, int n_in,
                              void* d_out, int out_size, void* d_ws, size_t ws_size,
                              hipStream_t stream) {
  const float* text  = (const float*)d_in[0];   // [768][512]
  const float* img4  = (const float*)d_in[1];   // [65536][512]
  const float* img8  = (const float*)d_in[2];   // [16384][512]
  const float* img12 = (const float*)d_in[3];   // [4096][512]
  const float* cache = (const float*)d_in[4];   // [430][512]
  const float* W     = (const float*)d_in[5];   // [512][1024]

  float* text_fine = (float*)d_out;             // [768][512]
  float* loss      = text_fine + 768 * 512;     // [1]
  float* newcache  = loss + 1;                  // [430][512]

  // -------- workspace carve (256B aligned) --------
  char* p = (char*)d_ws;
  auto carve = [&](size_t bytes) -> char* {
    char* r = p;
    p += (bytes + 255) & ~(size_t)255;
    return r;
  };
  __bf16* cache_hi  = (__bf16*)carve(448 * 512 * 2);
  __bf16* cache_lo  = (__bf16*)carve(448 * 512 * 2);
  __bf16* cacheT_hi = (__bf16*)carve(512 * 448 * 2);
  __bf16* cacheT_lo = (__bf16*)carve(512 * 448 * 2);
  __bf16* W_hi      = (__bf16*)carve(512 * 1024 * 2);
  __bf16* W_lo      = (__bf16*)carve(512 * 1024 * 2);
  float* rnorm_text = (float*)carve(768 * 4);
  float* rnorm4     = (float*)carve(65536 * 4);
  float* rnorm8     = (float*)carve(16384 * 4);
  float* rnorm12    = (float*)carve(4096 * 4);
  float* score_text = (float*)carve(768 * 448 * 4);
  float* P          = (float*)carve(768 * 448 * 4);
  float* Acat       = (float*)carve(768 * 1024 * 4);
  float* rowmax     = (float*)carve(65536 * 4);
  int*   rowarg     = (int*)carve(65536 * 4);
  unsigned* colmax  = (unsigned*)carve(3 * 448 * 4);
  float* sums       = (float*)carve(3 * 430 * 512 * 4);
  float* cnt        = (float*)carve(3 * 430 * 4);

  // allow >64KB dynamic LDS for the GEMM kernel (max need: 64*1024*4 + 2KB)
  (void)hipFuncSetAttribute((const void*)k_gemm,
                            hipFuncAttributeMaxDynamicSharedMemorySize, 266240);

  // -------- zero accumulators --------
  hipMemsetAsync(sums, 0, (size_t)3 * 430 * 512 * 4, stream);
  hipMemsetAsync(cnt, 0, 3 * 430 * 4, stream);
  hipMemsetAsync(colmax, 0, 3 * 448 * 4, stream);   // 0 == ordered-uint identity
  hipMemsetAsync(loss, 0, 4, stream);

  // -------- prep: bf16 hi/lo operand tables --------
  k_split<<<CDIV(448 * 512, 256), 256, 0, stream>>>(cache, 430, 512, cache_hi, cache_lo, 448, 512, 0);
  k_split<<<CDIV(512 * 448, 256), 256, 0, stream>>>(cache, 430, 512, cacheT_hi, cacheT_lo, 512, 448, 1);
  k_split<<<CDIV(512 * 1024, 256), 256, 0, stream>>>(W, 512, 1024, W_hi, W_lo, 512, 1024, 0);

  // -------- row norms --------
  k_rownorm<<<768 / 8, 256, 0, stream>>>(text, rnorm_text);
  k_rownorm<<<65536 / 8, 256, 0, stream>>>(img4, rnorm4);
  k_rownorm<<<16384 / 8, 256, 0, stream>>>(img8, rnorm8);
  k_rownorm<<<4096 / 8, 256, 0, stream>>>(img12, rnorm12);

  const size_t sh512  = (size_t)GEMM_ROWS * 512 * 4 + 448 * 4;
  const size_t sh448  = (size_t)GEMM_ROWS * 448 * 4 + 512 * 4;
  const size_t sh1024 = (size_t)GEMM_ROWS * 1024 * 4 + 512 * 4;

  // -------- read path --------
  // score_text = l2norm(text) @ cache^T   [768 x 430] (padded to 448)
  k_gemm<<<768 / GEMM_ROWS, GEMM_THREADS, sh512, stream>>>(
      text, 512, rnorm_text, cache_hi, cache_lo, 512, 512, 28, 430,
      score_text, 448, nullptr, 0.f, nullptr, nullptr, nullptr, 0);
  k_softmax<<<768 / 8, 256, 0, stream>>>(score_text, P);
  // fine = P @ cache  -> Acat[:, 512:1024]
  k_gemm<<<768 / GEMM_ROWS, GEMM_THREADS, sh448, stream>>>(
      P, 448, nullptr, cacheT_hi, cacheT_lo, 448, 448, 32, 512,
      Acat + 512, 1024, nullptr, 0.f, nullptr, nullptr, nullptr, 0);
  k_copy_text<<<CDIV(768 * 512, 256), 256, 0, stream>>>(text, Acat);
  // text_fine = 0.2 * (Acat @ W^T) + text
  k_gemm<<<768 / GEMM_ROWS, GEMM_THREADS, sh1024, stream>>>(
      Acat, 1024, nullptr, W_hi, W_lo, 1024, 1024, 32, 512,
      text_fine, 512, text, 0.2f, nullptr, nullptr, nullptr, 1);
  k_loss<<<768 / 8, 256, 0, stream>>>(text_fine, text, loss);

  // -------- write path (3 scales, fused score GEMM + reductions) --------
  const float* imgs[3] = {img4, img8, img12};
  const float* rns[3]  = {rnorm4, rnorm8, rnorm12};
  const int    Ns[3]   = {65536, 16384, 4096};
  for (int s = 0; s < 3; ++s) {
    k_gemm<<<Ns[s] / GEMM_ROWS, GEMM_THREADS, sh512, stream>>>(
        imgs[s], 512, rns[s], cache_hi, cache_lo, 512, 512, 28, 430,
        nullptr, 0, nullptr, 0.f, rowmax, rowarg, colmax + s * 448, 2);
    k_scatter<<<Ns[s] / 8, 256, 0, stream>>>(
        imgs[s], rns[s], rowmax, rowarg, colmax + s * 448,
        sums + (size_t)s * 430 * 512, cnt + s * 430, Ns[s]);
  }
  k_combine<<<CDIV(432, 8), 256, 0, stream>>>(cache, sums, cnt, newcache);
}
